// GRUDEncoder_20959440404675
// MI455X (gfx1250) — matrix operand order repaired and checked by reading, not compiled, not run on hardware
//
#include <hip/hip_runtime.h>

// ---------------- problem constants ----------------
constexpr int Bx = 128;   // batch
constexpr int Tx = 512;   // time steps
constexpr int Dx = 128;   // input dim
constexpr int Hx = 768;   // hidden
constexpr int K0 = Dx + Hx;   // 896  (layer0 comb width)
constexpr int K1 = Hx + Hx;   // 1536 (layer1 comb width)

typedef __attribute__((ext_vector_type(16))) __bf16 v16bf;
typedef __attribute__((ext_vector_type(8)))  __bf16 v8bf;
typedef __attribute__((ext_vector_type(8)))  float  v8f;

static __device__ __forceinline__ __bf16 f2bf(float f) {
    union { float f; unsigned u; } v; v.f = f;
    unsigned r = v.u + 0x7FFFu + ((v.u >> 16) & 1u);   // RNE
    unsigned short h = (unsigned short)(r >> 16);
    return __builtin_bit_cast(__bf16, h);
}

// fast transcendentals: v_exp_f32 + v_rcp_f32 instead of precise libm
static __device__ __forceinline__ float fast_sigmoid(float v) {
    return __builtin_amdgcn_rcpf(1.0f + __expf(-v));
}
static __device__ __forceinline__ float fast_tanh(float x) {
    float xc = fminf(fmaxf(x, -15.0f), 15.0f);      // keep exp finite
    float e  = __expf(2.0f * xc);
    return (e - 1.0f) * __builtin_amdgcn_rcpf(e + 1.0f);
}
static __device__ __forceinline__ float decay_gamma(float dl, float w, float b) {
    return __expf(-fmaxf(0.0f, dl * w + b));        // arg <= 0: safe
}

// ---------------- weight transpose + bf16 convert ----------------
// W [K,N] fp32 row-major  ->  Wt [(rowOff+n), k] bf16 row-major (ld = ldt)
__global__ void wtrans_kernel(const float* __restrict__ W, __bf16* __restrict__ Wt,
                              int K, int N, int rowOff, int ldt) {
    int idx = blockIdx.x * blockDim.x + threadIdx.x;
    if (idx >= K * N) return;
    int k = idx / N, n = idx % N;
    Wt[(size_t)(rowOff + n) * ldt + k] = f2bf(W[(size_t)k * N + n]);
}

// ---------------- t=0 initialization ----------------
// h0 starts at 0 => hdec0 = 0, comb0 h-part = 0; imputation for t=0; zero h1.
__global__ void init_kernel(const float* __restrict__ x, const float* __restrict__ mask,
                            const float* __restrict__ deltas,
                            const float* __restrict__ Wgx, const float* __restrict__ bgx,
                            const float* __restrict__ xmean,
                            float* __restrict__ h1, float* __restrict__ hdec0,
                            __bf16* __restrict__ comb0, __bf16* __restrict__ combr0) {
    int idx = blockIdx.x * blockDim.x + threadIdx.x;   // [0, Bx*Hx)
    int b = idx / Hx, j = idx % Hx;
    h1[idx] = 0.0f;
    hdec0[idx] = 0.0f;
    comb0[(size_t)b * K0 + Dx + j] = f2bf(0.0f);
    if (j < Dx) {
        float dl = deltas[(size_t)b * Tx + 0];
        float gx = decay_gamma(dl, Wgx[j], bgx[j]);
        size_t o = (size_t)b * Tx * Dx + j;            // t = 0
        float xv = x[o], mv = mask[o];
        float xi = mv * xv + (1.0f - mv) * (gx * xv + (1.0f - gx) * xmean[j]);
        __bf16 xb = f2bf(xi);
        comb0 [(size_t)b * K0 + j] = xb;
        combr0[(size_t)b * K0 + j] = xb;
    }
}

// ---------------- WMMA mainloop (bf16 -> f32 accum) ----------------
// A [128, K] bf16 row-major (lda = K); Bt [N, K] bf16 row-major (W transposed).
// One wave computes one 16x16 C tile; tileM = warp-in-block, tileN = blockIdx.x.
template <int K>
static __device__ __forceinline__ v8f wmma_tile(const __bf16* __restrict__ A,
                                                const __bf16* __restrict__ Bt,
                                                int m0, int n0, int lane) {
    v8f c = {};
    const int kA = (lane < 16) ? 0 : 16;   // A: lane<16 holds K {0..15}
    const int kB = (lane < 16) ? 0 : 16;   // B: lane<16 holds K {0..15}
    const __bf16* Ap = A + (size_t)(m0 + (lane & 15)) * K;
    const __bf16* Bp = Bt + (size_t)(n0 + (lane & 15)) * K;
#pragma unroll 2
    for (int kk = 0; kk < K; kk += 32) {
        v8bf alo = *(const v8bf*)(Ap + kk + kA);
        v8bf ahi = *(const v8bf*)(Ap + kk + kA + 8);
        v16bf a = __builtin_shufflevector(alo, ahi,
                    0,1,2,3,4,5,6,7,8,9,10,11,12,13,14,15);
        v16bf bm = *(const v16bf*)(Bp + kk + kB);
        c = __builtin_amdgcn_wmma_f32_16x16x32_bf16(
                false, a, false, bm, (short)0, c, false, false);
    }
    return c;
}

// ---------------- GEMM z|r : C[128, 2H] = comb @ [Wz|Wr]^T-packed ----------------
// cols [0,H)  -> z = sigmoid (fp32);  cols [H,2H) -> combr h-part = bf16(r * hdec)
template <int K>
__global__ void gemm_zr_kernel(const __bf16* __restrict__ A, const __bf16* __restrict__ Bt,
                               const float* __restrict__ bz, const float* __restrict__ br,
                               const float* __restrict__ hdec, float* __restrict__ z,
                               __bf16* __restrict__ combr, int xdim) {
    const int lane  = threadIdx.x & 31;
    const int tileM = threadIdx.x >> 5;     // 8 warps -> 8 M tiles (M = 128)
    const int tileN = blockIdx.x;           // 2H/16 = 96 blocks
    const int m0 = tileM * 16, n0 = tileN * 16;
    v8f c = wmma_tile<K>(A, Bt, m0, n0, lane);
    const int col   = n0 + (lane & 15);
    const int rbase = m0 + ((lane < 16) ? 0 : 8);
    const int CK    = xdim + Hx;
#pragma unroll
    for (int g = 0; g < 8; ++g) {
        int row = rbase + g;
        float v = c[g] + ((col < Hx) ? bz[col] : br[col - Hx]);
        float s = fast_sigmoid(v);
        if (col < Hx) {
            z[(size_t)row * Hx + col] = s;
        } else {
            int j = col - Hx;
            combr[(size_t)row * CK + xdim + j] = f2bf(s * hdec[(size_t)row * Hx + j]);
        }
    }
}

// ---------------- layer-0 h_tilde GEMM + update, fused with layer-1 prep(t) ----
__global__ void gemm_h0_kernel(const __bf16* __restrict__ A,  // combr0 [128, K0]
                               const __bf16* __restrict__ Bt, // Wh0t   [H, K0]
                               const float* __restrict__ bh, const float* __restrict__ z,
                               const float* __restrict__ hdec, float* __restrict__ h0,
                               __bf16* __restrict__ comb1, __bf16* __restrict__ combr1,
                               // fused layer-1 decay prep (step t):
                               const float* __restrict__ deltas, int t,
                               const float* __restrict__ Wgh1, const float* __restrict__ bgh1,
                               const float* __restrict__ h1_old, float* __restrict__ hdec1) {
    const int lane  = threadIdx.x & 31;
    const int tileM = threadIdx.x >> 5;
    const int tileN = blockIdx.x;           // H/16 = 48 blocks
    const int m0 = tileM * 16, n0 = tileN * 16;
    v8f c = wmma_tile<K0>(A, Bt, m0, n0, lane);
    const int col   = n0 + (lane & 15);
    const int rbase = m0 + ((lane < 16) ? 0 : 8);
    float wgh = Wgh1[col], bgh = bgh1[col], bhc = bh[col];
#pragma unroll
    for (int g = 0; g < 8; ++g) {
        int row = rbase + g;
        size_t o = (size_t)row * Hx + col;
        // GRU update for layer 0
        float ht = fast_tanh(c[g] + bhc);
        float zz = z[o];
        float hn = (1.0f - zz) * hdec[o] + zz * ht;
        h0[o] = hn;
        __bf16 hb = f2bf(hn);
        comb1 [(size_t)row * K1 + col] = hb;   // layer-1 x_imp (identity path)
        combr1[(size_t)row * K1 + col] = hb;
        // fused: layer-1 hidden decay for this step
        float dl = deltas[(size_t)row * Tx + t];
        float hd = decay_gamma(dl, wgh, bgh) * h1_old[o];
        hdec1[o] = hd;
        comb1[(size_t)row * K1 + Hx + col] = f2bf(hd);
    }
}

// ---- layer-1 h_tilde GEMM + update + output, fused with layer-0 prep(t+1) ----
__global__ void gemm_h1_kernel(const __bf16* __restrict__ A,  // combr1 [128, K1]
                               const __bf16* __restrict__ Bt, // Wh1t   [H, K1]
                               const float* __restrict__ bh, const float* __restrict__ z,
                               const float* __restrict__ hdec, float* __restrict__ h1,
                               float* __restrict__ out_ptr,   // d_out + t*H, row stride T*H
                               // fused layer-0 prep for step t+1:
                               int do_prep, int tn,
                               const float* __restrict__ deltas,
                               const float* __restrict__ Wgh0, const float* __restrict__ bgh0,
                               const float* __restrict__ h0, float* __restrict__ hdec0,
                               __bf16* __restrict__ comb0, __bf16* __restrict__ combr0,
                               const float* __restrict__ x, const float* __restrict__ mask,
                               const float* __restrict__ Wgx0, const float* __restrict__ bgx0,
                               const float* __restrict__ xmean0) {
    const int lane  = threadIdx.x & 31;
    const int tileM = threadIdx.x >> 5;
    const int tileN = blockIdx.x;           // H/16 = 48 blocks
    const int m0 = tileM * 16, n0 = tileN * 16;
    v8f c = wmma_tile<K1>(A, Bt, m0, n0, lane);
    const int col   = n0 + (lane & 15);
    const int rbase = m0 + ((lane < 16) ? 0 : 8);
    float wgh = Wgh0[col], bgh = bgh0[col], bhc = bh[col];
#pragma unroll
    for (int g = 0; g < 8; ++g) {
        int row = rbase + g;
        size_t o = (size_t)row * Hx + col;
        // GRU update for layer 1 + emit output
        float ht = fast_tanh(c[g] + bhc);
        float zz = z[o];
        float hn = (1.0f - zz) * hdec[o] + zz * ht;
        h1[o] = hn;
        out_ptr[(size_t)row * (size_t)(Tx * Hx) + col] = hn;
        // fused: layer-0 decay + imputation prep for next step
        if (do_prep) {
            float dl = deltas[(size_t)row * Tx + tn];
            float hd = decay_gamma(dl, wgh, bgh) * h0[o];
            hdec0[o] = hd;
            comb0[(size_t)row * K0 + Dx + col] = f2bf(hd);
            if (col < Dx) {
                float gx = decay_gamma(dl, Wgx0[col], bgx0[col]);
                size_t xo = ((size_t)row * Tx + tn) * Dx + col;
                float xv = x[xo], mv = mask[xo];
                float xi = mv * xv + (1.0f - mv) * (gx * xv + (1.0f - gx) * xmean0[col]);
                __bf16 xb = f2bf(xi);
                comb0 [(size_t)row * K0 + col] = xb;
                combr0[(size_t)row * K0 + col] = xb;
            }
        }
    }
}

// ---------------- final-hidden gather ----------------
__global__ void gather_kernel(const float* __restrict__ outs, const int* __restrict__ lens,
                              float* __restrict__ finals) {
    int idx = blockIdx.x * blockDim.x + threadIdx.x;   // Bx*Hx
    int b = idx / Hx, j = idx % Hx;
    int len = lens[b];
    if (len < 1) len = 1;
    if (len > Tx) len = Tx;
    finals[idx] = outs[(size_t)b * Tx * Hx + (size_t)(len - 1) * Hx + j];
}

// ---------------- host launcher ----------------
extern "C" void kernel_launch(void* const* d_in, const int* in_sizes, int n_in,
                              void* d_out, int out_size, void* d_ws, size_t ws_size,
                              hipStream_t stream) {
    const float* x      = (const float*)d_in[0];
    const float* mask   = (const float*)d_in[1];
    const float* deltas = (const float*)d_in[2];
    const int*   lens   = (const int*)d_in[3];

    // Param flatten order: insertion (Wz,bz,Wr,br,Wh,bh,Wgh,bgh,Wgx,bgx,x_mean)
    // vs jax sorted keys (Wgh,Wgx,Wh,Wr,Wz,bgh,bgx,bh,br,bz,x_mean). Detect by size.
    bool ins = (in_sizes[4] > 100000);
    int iWz, iBz, iWr, iBr, iWh, iBh, iWgh, iBgh, iWgx, iBgx, iXm;
    if (ins) { iWz=0; iBz=1; iWr=2; iBr=3; iWh=4; iBh=5; iWgh=6; iBgh=7; iWgx=8; iBgx=9; iXm=10; }
    else     { iWgh=0; iWgx=1; iWh=2; iWr=3; iWz=4; iBgh=5; iBgx=6; iBh=7; iBr=8; iBz=9; iXm=10; }
    const int b0 = 4, b1 = 15;
    auto P = [&](int base, int k) { return (const float*)d_in[base + k]; };

    // ---- workspace carve (256B aligned) ----
    char* ws = (char*)d_ws; size_t off = 0;
    auto alloc = [&](size_t bytes) -> void* {
        void* p = ws + off; off = (off + bytes + 255) & ~(size_t)255; return p;
    };
    __bf16* Wzr0 = (__bf16*)alloc((size_t)2 * Hx * K0 * 2);   // [1536, 896]
    __bf16* Wh0  = (__bf16*)alloc((size_t)Hx * K0 * 2);       // [768, 896]
    __bf16* Wzr1 = (__bf16*)alloc((size_t)2 * Hx * K1 * 2);   // [1536, 1536]
    __bf16* Wh1  = (__bf16*)alloc((size_t)Hx * K1 * 2);       // [768, 1536]
    __bf16* comb0  = (__bf16*)alloc((size_t)Bx * K0 * 2);
    __bf16* combr0 = (__bf16*)alloc((size_t)Bx * K0 * 2);
    __bf16* comb1  = (__bf16*)alloc((size_t)Bx * K1 * 2);
    __bf16* combr1 = (__bf16*)alloc((size_t)Bx * K1 * 2);
    float* h0    = (float*)alloc((size_t)Bx * Hx * 4);
    float* h1    = (float*)alloc((size_t)Bx * Hx * 4);
    float* hdec0 = (float*)alloc((size_t)Bx * Hx * 4);
    float* hdec1 = (float*)alloc((size_t)Bx * Hx * 4);
    float* z0    = (float*)alloc((size_t)Bx * Hx * 4);
    float* z1    = (float*)alloc((size_t)Bx * Hx * 4);

    const int TB = 256;
    // ---- weights -> bf16 transposed [N,K] ----
    {
        int n0e = K0 * Hx, g0 = (n0e + TB - 1) / TB;
        int n1e = K1 * Hx, g1 = (n1e + TB - 1) / TB;
        wtrans_kernel<<<g0, TB, 0, stream>>>(P(b0,iWz), Wzr0, K0, Hx, 0,  K0);
        wtrans_kernel<<<g0, TB, 0, stream>>>(P(b0,iWr), Wzr0, K0, Hx, Hx, K0);
        wtrans_kernel<<<g0, TB, 0, stream>>>(P(b0,iWh), Wh0,  K0, Hx, 0,  K0);
        wtrans_kernel<<<g1, TB, 0, stream>>>(P(b1,iWz), Wzr1, K1, Hx, 0,  K1);
        wtrans_kernel<<<g1, TB, 0, stream>>>(P(b1,iWr), Wzr1, K1, Hx, Hx, K1);
        wtrans_kernel<<<g1, TB, 0, stream>>>(P(b1,iWh), Wh1,  K1, Hx, 0,  K1);
    }

    const int gElem = (Bx * Hx) / TB;       // 384 blocks of 256
    const int gZR   = (2 * Hx) / 16;        // 96 blocks (8 M-tile warps each)
    const int gH    = Hx / 16;              // 48 blocks
    float* outs = (float*)d_out;

    // ---- t = 0 prep (h0 = h1 = 0) ----
    init_kernel<<<gElem, TB, 0, stream>>>(x, mask, deltas,
            P(b0,iWgx), P(b0,iBgx), P(b0,iXm), h1, hdec0, comb0, combr0);

    for (int t = 0; t < Tx; ++t) {
        // ----- layer 0 -----
        gemm_zr_kernel<K0><<<gZR, TB, 0, stream>>>(comb0, Wzr0,
                P(b0,iBz), P(b0,iBr), hdec0, z0, combr0, Dx);
        gemm_h0_kernel<<<gH, TB, 0, stream>>>(combr0, Wh0,
                P(b0,iBh), z0, hdec0, h0, comb1, combr1,
                deltas, t, P(b1,iWgh), P(b1,iBgh), h1, hdec1);
        // ----- layer 1 (epilogue also preps layer-0 for step t+1) -----
        gemm_zr_kernel<K1><<<gZR, TB, 0, stream>>>(comb1, Wzr1,
                P(b1,iBz), P(b1,iBr), hdec1, z1, combr1, Hx);
        gemm_h1_kernel<<<gH, TB, 0, stream>>>(combr1, Wh1,
                P(b1,iBh), z1, hdec1, h1, outs + (size_t)t * Hx,
                (t + 1 < Tx) ? 1 : 0, t + 1, deltas,
                P(b0,iWgh), P(b0,iBgh), h0, hdec0, comb0, combr0,
                x, mask, P(b0,iWgx), P(b0,iBgx), P(b0,iXm));
    }
    // ----- final hidden gather -----
    gather_kernel<<<gElem, TB, 0, stream>>>(outs, lens, outs + (size_t)Bx * Tx * Hx);
}
